// TrajectoryDecoder_49057116455152
// MI455X (gfx1250) — matrix operand order repaired and checked by reading, not compiled, not run on hardware
//
#include <hip/hip_runtime.h>
#include <hip/hip_bf16.h>
#include <cstdint>

#define D_MODEL 256
#define HIDDEN  512
#define OUT_DIM 32
#define N_TYPES 4
#define BLK_TOK 128                      // tokens per decode block (8 waves x 16)
#define W1ELTS (N_TYPES * D_MODEL * HIDDEN)   // 524288
#define W2ELTS (N_TYPES * HIDDEN * OUT_DIM)   // 65536

typedef __attribute__((ext_vector_type(16))) __bf16 v16bf;
typedef __attribute__((ext_vector_type(8)))  float  v8f;
typedef __attribute__((ext_vector_type(4)))  unsigned int u32x4;
typedef __attribute__((ext_vector_type(8)))  int i32x8;
typedef __attribute__((ext_vector_type(4)))  int i32x4;

union Frag16 {
  v16bf v;
  uint4 q[2];
  unsigned short u[16];
};

__device__ __forceinline__ unsigned short f2bf(float f) {
  unsigned u = __builtin_bit_cast(unsigned, f);
  u += 0x7FFFu + ((u >> 16) & 1u);           // round-to-nearest-even
  return (unsigned short)(u >> 16);
}

// TDM: DMA one contiguous 16KB panel (2048 x 8B) global -> LDS.
// D# group0: count=1 | lds_addr | global_addr[56:0] | type=2
// D# group1: data_size=8B, tensor_dim0=2048, tensor_dim1=1, tile_dim0=2048,
//            tile_dim1=1, dim0_stride=2048
__device__ __forceinline__ void tdm_load_16k(const unsigned short* gsrc, unsigned lds_off) {
  unsigned long long ga = (unsigned long long)(uintptr_t)gsrc;
  u32x4 g0 = { 1u, lds_off, (unsigned)ga,
               (unsigned)((ga >> 32) & 0x01FFFFFFu) | 0x80000000u };
  i32x8 g1 = { 3 << 16,                 // data_size = 8B
               (int)(2048u << 16),      // tensor_dim0[15:0] in [31:16]
               1 << 16,                 // tensor_dim0 hi = 0 | tensor_dim1 lo = 1
               (int)(2048u << 16),      // tensor_dim1 hi = 0 | tile_dim0 = 2048
               1,                       // tile_dim1 = 1, tile_dim2 = 0
               2048,                    // tensor_dim0_stride lo32
               0, 0 };
  i32x4 gz = { 0, 0, 0, 0 };
#if defined(__clang_major__) && __clang_major__ >= 23
  i32x8 gz8 = { 0, 0, 0, 0, 0, 0, 0, 0 };
  __builtin_amdgcn_tensor_load_to_lds(g0, g1, gz, gz, gz8, 0);
#else
  __builtin_amdgcn_tensor_load_to_lds(g0, g1, gz, gz, 0);
#endif
}

// ---------------- bucketing passes ----------------

__global__ void k_init(int* counts) {
  if (threadIdx.x < N_TYPES) counts[threadIdx.x] = 0;
}

__global__ void k_hist(const int* __restrict__ types, int* __restrict__ counts, int n) {
  __shared__ int lc[N_TYPES];
  if (threadIdx.x < N_TYPES) lc[threadIdx.x] = 0;
  __syncthreads();
  int i = blockIdx.x * blockDim.x + threadIdx.x;
  if (i < n) atomicAdd(&lc[types[i] & 3], 1);
  __syncthreads();
  if (threadIdx.x < N_TYPES) atomicAdd(&counts[threadIdx.x], lc[threadIdx.x]);
}

__global__ void k_offsets(const int* __restrict__ counts, int* __restrict__ p,
                          int* __restrict__ cursors) {
  if (threadIdx.x == 0) {
    int acc = 0;
    for (int t = 0; t < N_TYPES; ++t) {
      p[t] = acc;
      cursors[t] = acc;
      acc += (counts[t] + BLK_TOK - 1) / BLK_TOK * BLK_TOK;   // pad each segment
    }
    p[N_TYPES] = acc;
  }
}

__global__ void k_fill(int* __restrict__ idx, int n) {
  int i = blockIdx.x * blockDim.x + threadIdx.x;
  if (i < n) idx[i] = -1;
}

__global__ void k_scatter(const int* __restrict__ types, int* __restrict__ cursors,
                          int* __restrict__ idx, int n) {
  __shared__ int lc[N_TYPES], lbase[N_TYPES];
  if (threadIdx.x < N_TYPES) lc[threadIdx.x] = 0;
  __syncthreads();
  int i = blockIdx.x * blockDim.x + threadIdx.x;
  int t = 0, r = 0;
  const int valid = (i < n);
  if (valid) { t = types[i] & 3; r = atomicAdd(&lc[t], 1); }
  __syncthreads();
  if (threadIdx.x < N_TYPES)
    lbase[threadIdx.x] = atomicAdd(&cursors[threadIdx.x], lc[threadIdx.x]);
  __syncthreads();
  if (valid) idx[lbase[t] + r] = i;
}

// ---- weight conversion: fp32 -> bf16 laid out as WMMA B-fragment images ----
__global__ void k_convw(const float* __restrict__ W1, const float* __restrict__ W2,
                        unsigned short* __restrict__ w1b, unsigned short* __restrict__ w2b) {
  int d = blockIdx.x * blockDim.x + threadIdx.x;
  if (d < W1ELTS) {
    int h2 = d & 15, lane = (d >> 4) & 31, kc = (d >> 9) & 7, j = (d >> 12) & 31, t = d >> 17;
    int hi = lane >> 4;
    int k  = 32 * kc + 8 * hi + (h2 & 7) + 16 * (h2 >> 3);
    int nn = 16 * j + (lane & 15);
    w1b[d] = f2bf(W1[((size_t)t * D_MODEL + k) * HIDDEN + nn]);
  } else if (d < W1ELTS + W2ELTS) {
    int e = d - W1ELTS;
    int h2 = e & 15, lane = (e >> 4) & 31, kc2 = (e >> 9) & 15, n2 = (e >> 13) & 1, t = e >> 14;
    int hi = lane >> 4;
    int k  = 32 * kc2 + 8 * hi + (h2 & 7) + 16 * (h2 >> 3);
    int nn = 16 * n2 + (lane & 15);
    w2b[e] = f2bf(W2[((size_t)t * HIDDEN + k) * OUT_DIM + nn]);
  }
}

// ---------------- fused bucketed MLP decode ----------------

__global__ __launch_bounds__(256)
void k_decode(const float* __restrict__ x, const int* __restrict__ idx,
              const int* __restrict__ p,
              const unsigned short* __restrict__ w1b,
              const unsigned short* __restrict__ w2b,
              const float* __restrict__ b1, const float* __restrict__ b2,
              float* __restrict__ out) {
  __shared__ unsigned short sW1[2][2 * 8 * 32 * 16];  // 2 x 16 KB double-buffered W1 panel
  __shared__ unsigned short sBounce[8][32 * 16];      // 8 KB per-wave A2-fragment image

  const int tid  = threadIdx.x;
  const int wv   = tid >> 5;
  const int lane = tid & 31;
  const int hi   = lane >> 4;
  const int nn   = lane & 15;

  const int r0 = blockIdx.x * BLK_TOK;
  if (r0 >= p[N_TYPES]) return;
  int t = 0;
  while (t < N_TYPES - 1 && r0 >= p[t + 1]) ++t;

  const unsigned short* w1t = w1b + (size_t)t * (32 * 8 * 32 * 16);
  const unsigned short* w2t = w2b + (size_t)t * (2 * 16 * 32 * 16);

  // kick off TDM for panel 0 (hidden tiles j=0,1) into buffer 0
  if (wv == 0) tdm_load_16k(w1t, (unsigned)(uintptr_t)&sW1[0][0]);

  // Build A fragments for this wave's 16 tokens (A 16x32 bf16 layout, ISA 7.12.2)
  const int mytok = idx[r0 + wv * 16 + nn];
  Frag16 a[8];
  if (mytok >= 0) {
    const float4* rp = (const float4*)(x + (size_t)mytok * D_MODEL);
#pragma unroll
    for (int kc = 0; kc < 8; ++kc) {
      float4 fa0 = rp[8 * kc + 2 * hi];        // k = 32kc + 8hi + 0..3
      float4 fa1 = rp[8 * kc + 2 * hi + 1];    // k = 32kc + 8hi + 4..7
      float4 ga0 = rp[8 * kc + 2 * hi + 4];    // k = 32kc + 16 + 8hi + 0..3
      float4 ga1 = rp[8 * kc + 2 * hi + 5];    // k = 32kc + 16 + 8hi + 4..7
      a[kc].u[0]  = f2bf(fa0.x); a[kc].u[1]  = f2bf(fa0.y);
      a[kc].u[2]  = f2bf(fa0.z); a[kc].u[3]  = f2bf(fa0.w);
      a[kc].u[4]  = f2bf(fa1.x); a[kc].u[5]  = f2bf(fa1.y);
      a[kc].u[6]  = f2bf(fa1.z); a[kc].u[7]  = f2bf(fa1.w);
      a[kc].u[8]  = f2bf(ga0.x); a[kc].u[9]  = f2bf(ga0.y);
      a[kc].u[10] = f2bf(ga0.z); a[kc].u[11] = f2bf(ga0.w);
      a[kc].u[12] = f2bf(ga1.x); a[kc].u[13] = f2bf(ga1.y);
      a[kc].u[14] = f2bf(ga1.z); a[kc].u[15] = f2bf(ga1.w);
    }
  } else {
#pragma unroll
    for (int kc = 0; kc < 8; ++kc) {
      a[kc].q[0] = make_uint4(0, 0, 0, 0);
      a[kc].q[1] = make_uint4(0, 0, 0, 0);
    }
  }

  const v8f vzero = {0.f, 0.f, 0.f, 0.f, 0.f, 0.f, 0.f, 0.f};
  v8f acc2[2];
  acc2[0] = vzero;
  acc2[1] = vzero;

  for (int m = 0; m < 16; ++m) {             // 32-wide hidden k-chunks
    // prefetch next panel via TDM while current one is consumed
    if (wv == 0) {
      if (m < 15) {
        tdm_load_16k(w1t + (size_t)(m + 1) * 8192,
                     (unsigned)(uintptr_t)&sW1[(m + 1) & 1][0]);
        __builtin_amdgcn_s_wait_tensorcnt(1);   // panel m landed
      } else {
        __builtin_amdgcn_s_wait_tensorcnt(0);
      }
    }
    __syncthreads();                           // panel m visible to all waves
    const unsigned short* pan = &sW1[m & 1][0];

#pragma unroll
    for (int jsel = 0; jsel < 2; ++jsel) {
      v8f acc = vzero;
#pragma unroll
      for (int kc = 0; kc < 8; ++kc) {
        Frag16 b;
        const uint4* bp = (const uint4*)&pan[((jsel * 8 + kc) * 32 + lane) * 16];
        b.q[0] = bp[0];
        b.q[1] = bp[1];
        acc = __builtin_amdgcn_wmma_f32_16x16x32_bf16(false, a[kc].v, false, b.v,
                                                      (short)0, acc, false, false);
      }
      // bias + ReLU, scatter C-layout (M=v+8hi, N=nn) into A2 fragment image
      const float b1v = b1[t * HIDDEN + 16 * (2 * m + jsel) + nn];
      const int K2 = jsel * 16 + nn;
      const int Lt_hi = 16 * ((K2 >> 3) & 1);
      const int h2 = 2 * (((K2 >> 1) & 3) + 4 * ((K2 >> 4) & 1)) + (K2 & 1);
#pragma unroll
      for (int v = 0; v < 8; ++v) {
        float hval = fmaxf(acc[v] + b1v, 0.0f);
        int tokl = v + 8 * hi;
        sBounce[wv][(tokl + Lt_hi) * 16 + h2] = f2bf(hval);
      }
    }

    asm volatile("s_wait_dscnt 0" ::: "memory");   // intra-wave LDS bounce RAW

    Frag16 a2;
    {
      const uint4* ap = (const uint4*)&sBounce[wv][lane * 16];
      a2.q[0] = ap[0];
      a2.q[1] = ap[1];
    }
    // B2 fragments straight from L2-resident pre-swizzled W2
#pragma unroll
    for (int n2 = 0; n2 < 2; ++n2) {
      Frag16 b;
      const uint4* bp = (const uint4*)&w2t[((n2 * 16 + m) * 32 + lane) * 16];
      b.q[0] = bp[0];
      b.q[1] = bp[1];
      acc2[n2] = __builtin_amdgcn_wmma_f32_16x16x32_bf16(false, a2.v, false, b.v,
                                                         (short)0, acc2[n2], false, false);
    }
    __syncthreads();     // all reads of buffer (m&1) done before TDM overwrites it
  }

  // y = acc2 + b2, scatter to original token order
#pragma unroll
  for (int n2 = 0; n2 < 2; ++n2) {
    const float b2v = b2[t * OUT_DIM + 16 * n2 + nn];
#pragma unroll
    for (int v = 0; v < 8; ++v) {
      int tok = idx[r0 + wv * 16 + v + 8 * hi];
      if (tok >= 0) out[(size_t)tok * OUT_DIM + 16 * n2 + nn] = acc2[n2][v] + b2v;
    }
  }
}

// ---------------- host launcher ----------------

extern "C" void kernel_launch(void* const* d_in, const int* in_sizes, int n_in,
                              void* d_out, int out_size, void* d_ws, size_t ws_size,
                              hipStream_t stream) {
  const float* x     = (const float*)d_in[0];
  const int*   types = (const int*)d_in[1];
  const float* W1    = (const float*)d_in[2];
  const float* b1    = (const float*)d_in[3];
  const float* W2    = (const float*)d_in[4];
  const float* b2    = (const float*)d_in[5];
  float* out = (float*)d_out;
  const int n = in_sizes[1];

  // workspace layout
  int* counts  = (int*)d_ws;           // [4]
  int* p       = counts + 8;           // [5] padded segment starts
  int* cursors = counts + 16;          // [4]
  int* idx     = counts + 32;          // [n + 4*BLK_TOK]
  const int idx_cap = n + N_TYPES * BLK_TOK;
  unsigned short* w1b = (unsigned short*)(idx + ((idx_cap + 3) & ~3));
  unsigned short* w2b = w1b + W1ELTS;

  k_init<<<1, 64, 0, stream>>>(counts);
  k_hist<<<(n + 255) / 256, 256, 0, stream>>>(types, counts, n);
  k_offsets<<<1, 32, 0, stream>>>(counts, p, cursors);
  k_fill<<<(idx_cap + 255) / 256, 256, 0, stream>>>(idx, idx_cap);
  k_scatter<<<(n + 255) / 256, 256, 0, stream>>>(types, cursors, idx, n);
  k_convw<<<(W1ELTS + W2ELTS + 255) / 256, 256, 0, stream>>>(W1, W2, w1b, w2b);
  k_decode<<<(n + BLK_TOK - 1) / BLK_TOK + N_TYPES, 256, 0, stream>>>(
      x, idx, p, w1b, w2b, b1, b2, out);
}
